// MSLoss_44573170598306
// MI455X (gfx1250) — matrix operand order repaired and checked by reading, not compiled, not run on hardware
//
#include <hip/hip_runtime.h>
#include <hip/hip_bf16.h>
#include <math.h>

// Problem constants (from reference)
#define NROWS 4096
#define DIM   512
static constexpr float ALPHA = 10.0f;
static constexpr float BETA  = 2.0f;
static constexpr float BASE_ = 0.5f;

typedef __attribute__((ext_vector_type(16))) __bf16 v16bf;
typedef __attribute__((ext_vector_type(8)))  float  v8f;

union Frag { v16bf v; uint4 q[2]; };

static __device__ inline v8f wmma_bf16(v16bf a, v16bf b, v8f c) {
    // (neg_a, A, neg_b, B, c_mod, C, reuse_a, reuse_b)
    return __builtin_amdgcn_wmma_f32_16x16x32_bf16(false, a, false, b, (short)0, c, false, false);
}

// A-matrix 16x32 bf16 fragment (ISA 7.12.2): lane l<16 -> M=l, K = {0..7, 16..23};
// lane l>=16 -> M=l-16, K = {8..15, 24..31}. Two 16B chunks per lane.
static __device__ inline v16bf load_a_frag(const unsigned short* Xb, int row, int k0, int lane) {
    const int half = (lane & 16) ? 8 : 0;
    const unsigned short* p = Xb + (size_t)row * DIM + k0 + half;
    Frag f;
    f.q[0] = *reinterpret_cast<const uint4*>(p);
    f.q[1] = *reinterpret_cast<const uint4*>(p + 16);
    return f.v;
}

// B-matrix 32x16 bf16 fragment: lane n<16 -> N=n, K = {0..7, 16..23}; lane n>=16 -> N=n-16, K = {8..15, 24..31}.
// B = X^T, so column n of B is row n of X: two 16B chunks per lane, 16 elements apart.
static __device__ inline v16bf load_b_frag(const unsigned short* Xb, int col, int k0, int lane) {
    const int khalf = (lane & 16) ? 8 : 0;
    const unsigned short* p = Xb + (size_t)col * DIM + k0 + khalf;
    Frag f;
    f.q[0] = reinterpret_cast<const uint4*>(p)[0];
    f.q[1] = reinterpret_cast<const uint4*>(p)[2];
    return f.v;
}

// 16x16 output tile of sim = X X^T via bf16x3 split (hi*hi + hi*lo + lo*hi), fp32 accum.
static __device__ inline v8f tile_gemm(const unsigned short* Xhi, const unsigned short* Xlo,
                                       int r0, int c0, int lane) {
    v8f acc = {0.f, 0.f, 0.f, 0.f, 0.f, 0.f, 0.f, 0.f};
    const int mrow = r0 + (lane & 15);
    const int ncol = c0 + (lane & 15);
    #pragma unroll 4
    for (int k0 = 0; k0 < DIM; k0 += 32) {
        v16bf ah = load_a_frag(Xhi, mrow, k0, lane);
        v16bf al = load_a_frag(Xlo, mrow, k0, lane);
        v16bf bh = load_b_frag(Xhi, ncol, k0, lane);
        v16bf bl = load_b_frag(Xlo, ncol, k0, lane);
        acc = wmma_bf16(ah, bh, acc);
        acc = wmma_bf16(ah, bl, acc);
        acc = wmma_bf16(al, bh, acc);
    }
    return acc;
}

// Round-to-nearest-even f32 -> bf16 bits
static __device__ inline unsigned short f2bf_rne(float f) {
    unsigned int u = __float_as_uint(f);
    return (unsigned short)((u + 0x7FFFu + ((u >> 16) & 1u)) >> 16);
}

// ---------------- Kernel 0: split fp32 X into bf16 hi + bf16 lo ----------------
__global__ __launch_bounds__(256) void split_kernel(const float* __restrict__ X,
                                                    unsigned short* __restrict__ Xhi,
                                                    unsigned short* __restrict__ Xlo) {
    int i = blockIdx.x * 256 + threadIdx.x;
    if (i < NROWS * DIM) {
        float f = X[i];
        unsigned short hs = f2bf_rne(f);
        float hf = __uint_as_float(((unsigned int)hs) << 16);
        unsigned short ls = f2bf_rne(f - hf);
        Xhi[i] = hs;
        Xlo[i] = ls;
    }
}

// ---------------- Kernel 1: per-row min_pos / max_neg (fused GEMM epilogue) ----------------
// Block b owns rows [16b, 16b+16); 8 waves sweep the 256 column tiles.
__global__ __launch_bounds__(256) void pass1_kernel(const unsigned short* __restrict__ Xhi,
                                                    const unsigned short* __restrict__ Xlo,
                                                    const int* __restrict__ targets,
                                                    float* __restrict__ min_pos,
                                                    float* __restrict__ max_neg) {
    const int lane = threadIdx.x & 31;
    const int wave = threadIdx.x >> 5;
    const int r0 = blockIdx.x * 16;
    const int rbase = r0 + ((lane & 16) ? 8 : 0);

    float mn[8], mx[8];
    int trow[8];
    #pragma unroll
    for (int j = 0; j < 8; ++j) {
        mn[j] = __builtin_inff();
        mx[j] = -__builtin_inff();
        trow[j] = targets[rbase + j];
    }

    for (int ct = wave; ct < NROWS / 16; ct += 8) {
        const int c0 = ct * 16;
        v8f acc = tile_gemm(Xhi, Xlo, r0, c0, lane);
        const int col = c0 + (lane & 15);
        const int tc = targets[col];
        #pragma unroll
        for (int j = 0; j < 8; ++j) {
            const float s = acc[j];
            const int row = rbase + j;
            if (tc == trow[j]) {
                if (row != col && s < 1.0f) mn[j] = fminf(mn[j], s);
            } else {
                mx[j] = fmaxf(mx[j], s);
            }
        }
    }
    // reduce across the 16 lanes of each half (xor masks 1..8 stay within a half)
    #pragma unroll
    for (int j = 0; j < 8; ++j) {
        #pragma unroll
        for (int m = 1; m <= 8; m <<= 1) {
            mn[j] = fminf(mn[j], __shfl_xor(mn[j], m, 32));
            mx[j] = fmaxf(mx[j], __shfl_xor(mx[j], m, 32));
        }
    }
    __shared__ float smin[8][16];
    __shared__ float smax[8][16];
    if ((lane & 15) == 0) {
        const int rloc = (lane & 16) ? 8 : 0;
        #pragma unroll
        for (int j = 0; j < 8; ++j) { smin[wave][rloc + j] = mn[j]; smax[wave][rloc + j] = mx[j]; }
    }
    __syncthreads();
    if (threadIdx.x < 16) {
        float a = __builtin_inff(), b = -__builtin_inff();
        #pragma unroll
        for (int w = 0; w < 8; ++w) { a = fminf(a, smin[w][threadIdx.x]); b = fmaxf(b, smax[w][threadIdx.x]); }
        min_pos[r0 + threadIdx.x] = a;
        max_neg[r0 + threadIdx.x] = b;
    }
}

// ---------------- Kernel 2: hard-mined exp sums + row loss (fused GEMM epilogue) ----------------
__global__ __launch_bounds__(256) void pass2_kernel(const unsigned short* __restrict__ Xhi,
                                                    const unsigned short* __restrict__ Xlo,
                                                    const int* __restrict__ targets,
                                                    const float* __restrict__ min_pos,
                                                    const float* __restrict__ max_neg,
                                                    const float* __restrict__ marginp,
                                                    float* __restrict__ loss_part,
                                                    float* __restrict__ inv_part) {
    const float margin = *marginp;
    const int lane = threadIdx.x & 31;
    const int wave = threadIdx.x >> 5;
    const int r0 = blockIdx.x * 16;
    const int rbase = r0 + ((lane & 16) ? 8 : 0);

    float posS[8], negS[8], posC[8], negC[8], mp[8], mxn[8];
    int trow[8];
    #pragma unroll
    for (int j = 0; j < 8; ++j) {
        posS[j] = negS[j] = posC[j] = negC[j] = 0.f;
        mp[j]  = min_pos[rbase + j];
        mxn[j] = max_neg[rbase + j];
        trow[j] = targets[rbase + j];
    }

    for (int ct = wave; ct < NROWS / 16; ct += 8) {
        const int c0 = ct * 16;
        v8f acc = tile_gemm(Xhi, Xlo, r0, c0, lane);
        const int col = c0 + (lane & 15);
        const int tc = targets[col];
        #pragma unroll
        for (int j = 0; j < 8; ++j) {
            const float s = acc[j];
            const int row = rbase + j;
            if (tc != trow[j]) {
                if (s + margin - mp[j] > 0.f) {           // sel_neg
                    negS[j] += __expf(ALPHA * (s - BASE_));
                    negC[j] += 1.f;
                }
            } else if (row != col && s < 1.0f) {
                if (mxn[j] - s + margin > 0.f) {          // sel_pos
                    posS[j] += __expf(-BETA * (s - BASE_));
                    posC[j] += 1.f;
                }
            }
        }
    }
    #pragma unroll
    for (int j = 0; j < 8; ++j) {
        #pragma unroll
        for (int m = 1; m <= 8; m <<= 1) {
            posS[j] += __shfl_xor(posS[j], m, 32);
            negS[j] += __shfl_xor(negS[j], m, 32);
            posC[j] += __shfl_xor(posC[j], m, 32);
            negC[j] += __shfl_xor(negC[j], m, 32);
        }
    }
    __shared__ float sPS[8][16], sNS[8][16], sPC[8][16], sNC[8][16];
    if ((lane & 15) == 0) {
        const int rloc = (lane & 16) ? 8 : 0;
        #pragma unroll
        for (int j = 0; j < 8; ++j) {
            sPS[wave][rloc + j] = posS[j];
            sNS[wave][rloc + j] = negS[j];
            sPC[wave][rloc + j] = posC[j];
            sNC[wave][rloc + j] = negC[j];
        }
    }
    __syncthreads();
    __shared__ float rowLoss[16];
    __shared__ float rowInv[16];
    if (threadIdx.x < 16) {
        float ps = 0.f, ns = 0.f, pc = 0.f, nc = 0.f;
        #pragma unroll
        for (int w = 0; w < 8; ++w) {
            ps += sPS[w][threadIdx.x]; ns += sNS[w][threadIdx.x];
            pc += sPC[w][threadIdx.x]; nc += sNC[w][threadIdx.x];
        }
        const bool valid = (pc > 0.f) && (nc > 0.f);
        rowLoss[threadIdx.x] = valid ? ((2.0f / BETA) * log1pf(ps) + (2.0f / ALPHA) * log1pf(ns)) : 0.f;
        rowInv[threadIdx.x]  = valid ? 0.f : 1.f;
    }
    __syncthreads();
    if (threadIdx.x == 0) {
        float l = 0.f, iv = 0.f;
        #pragma unroll
        for (int t = 0; t < 16; ++t) { l += rowLoss[t]; iv += rowInv[t]; }
        loss_part[blockIdx.x] = l;   // per-block slot: deterministic, no atomics
        inv_part[blockIdx.x]  = iv;
    }
}

// ---------------- Kernel 3: last-row stats (exact fp32) + final reduction ----------------
__global__ __launch_bounds__(256) void finalize_kernel(const float* __restrict__ X,
                                                       const int* __restrict__ targets,
                                                       const float* __restrict__ loss_part,
                                                       const float* __restrict__ inv_part,
                                                       float* __restrict__ out) {
    __shared__ float xlast[DIM];
    __shared__ float red[256];
    const int tid = threadIdx.x;
    for (int i = tid; i < DIM; i += 256) xlast[i] = X[(size_t)(NROWS - 1) * DIM + i];
    __syncthreads();

    const int tlast = targets[NROWS - 1];
    float sp = 0.f, cp = 0.f, sn = 0.f, cn = 0.f;
    for (int j = tid; j < NROWS; j += 256) {
        const float* xr = X + (size_t)j * DIM;
        float d = 0.f;
        #pragma unroll 8
        for (int k = 0; k < DIM; ++k) d = fmaf(xr[k], xlast[k], d);
        if (targets[j] == tlast) {
            if (d < 1.0f) { sp += d; cp += 1.f; }   // pos_mask (self excluded by sim<1)
        } else {
            sn += d; cn += 1.f;
        }
    }
    const float lp = loss_part[tid];   // exactly 256 partials
    const float ip = inv_part[tid];

    auto blockSum = [&](float v) -> float {
        red[tid] = v; __syncthreads();
        for (int s = 128; s > 0; s >>= 1) {
            if (tid < s) red[tid] += red[tid + s];
            __syncthreads();
        }
        float r = red[0]; __syncthreads();
        return r;
    };
    const float Sl  = blockSum(lp);
    const float Si  = blockSum(ip);
    const float Ssp = blockSum(sp);
    const float Scp = blockSum(cp);
    const float Ssn = blockSum(sn);
    const float Scn = blockSum(cn);
    if (tid == 0) {
        out[0] = Sl / (float)NROWS;                 // loss
        out[1] = Si / (float)NROWS;                 // prec
        out[2] = Ssp / fmaxf(Scp, 1.0f);            // mean_pos_sim (last row)
        out[3] = Ssn / fmaxf(Scn, 1.0f);            // mean_neg_sim (last row)
    }
}

extern "C" void kernel_launch(void* const* d_in, const int* in_sizes, int n_in,
                              void* d_out, int out_size, void* d_ws, size_t ws_size,
                              hipStream_t stream) {
    (void)in_sizes; (void)n_in; (void)out_size; (void)ws_size;
    const float* X       = (const float*)d_in[0];
    const int*   targets = (const int*)d_in[1];
    const float* margin  = (const float*)d_in[2];
    float* out = (float*)d_out;

    // Workspace layout (all 16B+ aligned): ~8 MB total
    char* ws = (char*)d_ws;
    unsigned short* Xhi = (unsigned short*)ws;                                   // 4 MB
    unsigned short* Xlo = (unsigned short*)(ws + (size_t)NROWS * DIM * 2);       // 4 MB
    float* min_pos   = (float*)(ws + (size_t)NROWS * DIM * 4);                   // 16 KB
    float* max_neg   = min_pos + NROWS;                                          // 16 KB
    float* loss_part = max_neg + NROWS;                                          // 1 KB
    float* inv_part  = loss_part + 256;                                          // 1 KB

    split_kernel<<<(NROWS * DIM + 255) / 256, 256, 0, stream>>>(X, Xhi, Xlo);
    pass1_kernel<<<NROWS / 16, 256, 0, stream>>>(Xhi, Xlo, targets, min_pos, max_neg);
    pass2_kernel<<<NROWS / 16, 256, 0, stream>>>(Xhi, Xlo, targets, min_pos, max_neg,
                                                 margin, loss_part, inv_part);
    finalize_kernel<<<1, 256, 0, stream>>>(X, targets, loss_part, inv_part, out);
}
